// HMHSA_Q_4234837754287
// MI455X (gfx1250) — matrix operand order repaired and checked by reading, not compiled, not run on hardware
//
#include <hip/hip_runtime.h>
#include <hip/hip_bf16.h>
#include <math.h>

typedef __attribute__((ext_vector_type(16))) _Float16 v16h;
typedef __attribute__((ext_vector_type(8)))  _Float16 v8h;
typedef __attribute__((ext_vector_type(8)))  float    v8f;

#define N_NODES  4096
#define IN_FEATS 256
#define EMBED    256
#define HEADS    8
#define HEAD_DIM 32
#define SCALING  0.17677669529663687f   // 32^-0.5

// ---------- small helpers -----------------------------------------------

__device__ __forceinline__ v16h ld_pair(const _Float16* p0, const _Float16* p1) {
  v8h a = *(const v8h*)p0;
  v8h b = *(const v8h*)p1;
  v16h r;
#pragma unroll
  for (int i = 0; i < 8; ++i) { r[i] = a[i]; r[8 + i] = b[i]; }
  return r;
}
__device__ __forceinline__ v16h ld_contig16(const _Float16* p) {
  return ld_pair(p, p + 8);
}

// ---------- f32 -> f16 convert ------------------------------------------

__global__ void cvt_f16_kernel(const float* __restrict__ src,
                               _Float16* __restrict__ dst, int n) {
  int i = blockIdx.x * blockDim.x + threadIdx.x;
  if (i < n) dst[i] = (_Float16)src[i];
}

// ---------- adj -> bitmask (wave32 ballot) ------------------------------
// adjbits[q*128 + w] bit b  <=>  adj[q][w*32+b] > 0

__global__ void pack_adj_kernel(const int* __restrict__ adj,
                                unsigned* __restrict__ bits) {
  int wid  = blockIdx.x * (blockDim.x >> 5) + (threadIdx.x >> 5);
  int lane = threadIdx.x & 31;
  int q = wid >> 7;            // 128 words per row
  int w = wid & 127;
  int k = w * 32 + lane;
  unsigned long long b = __ballot(adj[(size_t)q * N_NODES + k] > 0);
  if (lane == 0) bits[(size_t)q * 128 + w] = (unsigned)b;
}

// ---------- projection GEMM: Y = x @ W^T + b ----------------------------
// grid.x = 4096/16 row tiles, grid.y = 0:Q 1:K 2:V, 8 waves cover 256 cols.

__global__ void __launch_bounds__(256)
proj_kernel(const _Float16* __restrict__ xh,
            const _Float16* __restrict__ Wqh, const _Float16* __restrict__ Wkh,
            const _Float16* __restrict__ Wvh,
            const float* __restrict__ bq, const float* __restrict__ bk,
            const float* __restrict__ bv,
            _Float16* __restrict__ Qh, _Float16* __restrict__ Kh,
            float* __restrict__ Vf) {
  const int lane  = threadIdx.x & 31;
  const int wave  = threadIdx.x >> 5;
  const int m0    = blockIdx.x * 16;
  const int e0    = wave * 32;
  const int which = blockIdx.y;
  const _Float16* W = (which == 0) ? Wqh : (which == 1) ? Wkh : Wvh;
  const float* bias = (which == 0) ? bq  : (which == 1) ? bk  : bv;

  const int n16  = lane & 15;
  const int hi   = lane >> 4;
  const int loA  = hi ? 8 : 0;    // unused; A's K chunk is also chosen by selB
  const int selB = hi ? 16 : 0;   // B operand K-chunk select

  v8f acc0 = {}; v8f acc1 = {};
#pragma unroll
  for (int kc = 0; kc < IN_FEATS; kc += 32) {
    const _Float16* ap = xh + (size_t)(m0 + n16) * IN_FEATS + kc;
    v16h A  = ld_contig16(ap + selB);
    v16h B0 = ld_contig16(W + (size_t)(e0 + n16)      * IN_FEATS + kc + selB);
    v16h B1 = ld_contig16(W + (size_t)(e0 + 16 + n16) * IN_FEATS + kc + selB);
    acc0 = __builtin_amdgcn_wmma_f32_16x16x32_f16(false, A, false, B0, (short)0, acc0, false, false);
    acc1 = __builtin_amdgcn_wmma_f32_16x16x32_f16(false, A, false, B1, (short)0, acc1, false, false);
  }
  float b0 = bias[e0 + n16], b1 = bias[e0 + 16 + n16];
#pragma unroll
  for (int r = 0; r < 8; ++r) {
    int row  = m0 + r + 8 * hi;
    float y0 = acc0[r] + b0, y1 = acc1[r] + b1;
    if (which == 2) {
      Vf[(size_t)row * EMBED + e0 + n16]      = y0;
      Vf[(size_t)row * EMBED + e0 + 16 + n16] = y1;
    } else {
      _Float16* D = (which == 0) ? Qh : Kh;
      D[(size_t)row * EMBED + e0 + n16]      = (_Float16)y0;
      D[(size_t)row * EMBED + e0 + 16 + n16] = (_Float16)y1;
    }
  }
}

// ---------- softmax(V) over head_dim, stored TRANSPOSED [h][d][n] f16 ---

__global__ void svt_kernel(const float* __restrict__ Vf,
                           _Float16* __restrict__ SVT) {
  int tid = blockIdx.x * blockDim.x + threadIdx.x;
  if (tid >= N_NODES * HEADS) return;
  int n = tid >> 3, h = tid & 7;
  const float* v = Vf + (size_t)n * EMBED + h * HEAD_DIM;
  float buf[HEAD_DIM];
  float m = -1e30f;
#pragma unroll
  for (int d = 0; d < HEAD_DIM; ++d) { buf[d] = v[d]; m = fmaxf(m, buf[d]); }
  float s = 0.f;
#pragma unroll
  for (int d = 0; d < HEAD_DIM; ++d) { buf[d] = __expf(buf[d] - m); s += buf[d]; }
  float inv = 1.0f / s;
  _Float16* o = SVT + (size_t)h * HEAD_DIM * N_NODES + n;
#pragma unroll
  for (int d = 0; d < HEAD_DIM; ++d)
    o[(size_t)d * N_NODES] = (_Float16)(buf[d] * inv);
}

// ---------- fused masked-attention (flash style, online softmax) --------
// grid = (32, 8 heads), block = 256 (8 waves); each wave owns 16 query rows.
// K / softmax(V)^T tiles are staged into LDS once per block with async
// global->LDS copies (double-buffered, ASYNCcnt), removing the 8x redundant
// per-wave global reads.

__global__ void __launch_bounds__(256)
attn_kernel(const _Float16* __restrict__ Qh, const _Float16* __restrict__ Kh,
            const _Float16* __restrict__ SVT, const float* __restrict__ Vf,
            const unsigned* __restrict__ adjbits, float* __restrict__ out) {
  // per buffer: halves [0..1023] = K tile [key][d], [1024..2047] = SVT tile [d][k]
  __shared__ alignas(16) _Float16 tiles[2][2048];
  __shared__ alignas(16) _Float16 plds[8 * 16 * 32];   // per-wave P staging

  const int lane = threadIdx.x & 31;
  const int wave = threadIdx.x >> 5;
  const int h    = blockIdx.y;
  const int q0   = blockIdx.x * 128 + wave * 16;
  const int n16  = lane & 15;
  const int hi   = lane >> 4;
  const int loA  = hi ? 8 : 0;
  const int selB = hi ? 16 : 0;

  // Q tile, A-operand layout
  const _Float16* qp = Qh + (size_t)(q0 + n16) * EMBED + h * HEAD_DIM;
  const v16h Aq = ld_contig16(qp + selB);

  const _Float16* svt_h = SVT + (size_t)h * HEAD_DIM * N_NODES;

  // async tile prefetch: thread t moves one 16B chunk; 256 threads = 4KB
  // (one global_load_async_to_lds_b128 per wave per iteration)
  const int t = threadIdx.x;
  auto issue_tile = [&](int buf, int k0) {
    unsigned lds_off =
        (unsigned)(uintptr_t)(const void*)&tiles[buf][0] + (unsigned)t * 16u;
    const char* g;
    if (t < 128) {        // K tile: key = t>>2, 16B part = t&3
      g = (const char*)(Kh + ((size_t)(k0 + (t >> 2)) * EMBED + h * HEAD_DIM))
          + (t & 3) * 16;
    } else {              // SVT tile: d = (t-128)>>2
      int t2 = t - 128;
      g = (const char*)(svt_h + (size_t)(t2 >> 2) * N_NODES + k0)
          + (t2 & 3) * 16;
    }
    asm volatile("global_load_async_to_lds_b128 %0, %1, off"
                 :: "v"(lds_off), "v"(g) : "memory");
  };

  const unsigned* rowbits[8];
#pragma unroll
  for (int r = 0; r < 8; ++r)
    rowbits[r] = adjbits + (size_t)(q0 + r + 8 * hi) * (N_NODES / 32);

  float m[8], lsum[8];
#pragma unroll
  for (int r = 0; r < 8; ++r) { m[r] = -1e30f; lsum[r] = 0.f; }
  v8f c0 = {}; v8f c1 = {};
  const v8f zacc = {};
  _Float16* pw = plds + wave * (16 * 32);

  issue_tile(0, 0);
  int buf = 0;

  for (int k0 = 0; k0 < N_NODES; k0 += 32) {
    // own slice landed; barrier => whole tile landed AND everyone is done
    // reading the other buffer, so it is safe to prefetch into it.
    asm volatile("s_wait_asynccnt 0x0" ::: "memory");
    __syncthreads();
    int kn = (k0 + 32 < N_NODES) ? (k0 + 32) : 0;   // last issue is harmless
    issue_tile(buf ^ 1, kn);

    const _Float16* Kb = &tiles[buf][0];
    const _Float16* Sb = &tiles[buf][1024];

    // scores: two 16x16 tiles (keys k0..k0+15, k0+16..k0+31)
    v16h Bk0 = ld_contig16(Kb + (size_t)(n16)      * 32 + selB);
    v16h Bk1 = ld_contig16(Kb + (size_t)(16 + n16) * 32 + selB);
    v8f s0 = __builtin_amdgcn_wmma_f32_16x16x32_f16(false, Aq, false, Bk0, (short)0, zacc, false, false);
    v8f s1 = __builtin_amdgcn_wmma_f32_16x16x32_f16(false, Aq, false, Bk1, (short)0, zacc, false, false);

    // softmax(V)^T block as B operands
    v16h Bs0 = ld_contig16(Sb + (size_t)(n16)      * 32 + selB);
    v16h Bs1 = ld_contig16(Sb + (size_t)(16 + n16) * 32 + selB);

    const int wb = k0 >> 5;
    float p0[8], p1[8], alpha[8];
#pragma unroll
    for (int r = 0; r < 8; ++r) {
      unsigned w = rowbits[r][wb];
      float a0 = s0[r], a1 = s1[r];
      // leaky_relu(0.01) then scale, then additive mask
      a0 = (fmaxf(a0, 0.f) + 0.01f * fminf(a0, 0.f)) * SCALING;
      a1 = (fmaxf(a1, 0.f) + 0.01f * fminf(a1, 0.f)) * SCALING;
      a0 = ((w >> n16)        & 1u) ? a0 : -1e30f;
      a1 = ((w >> (16 + n16)) & 1u) ? a1 : -1e30f;
      // row max across the 16 lanes holding this row
      float tm = fmaxf(a0, a1);
      tm = fmaxf(tm, __shfl_xor(tm, 1, 32));
      tm = fmaxf(tm, __shfl_xor(tm, 2, 32));
      tm = fmaxf(tm, __shfl_xor(tm, 4, 32));
      tm = fmaxf(tm, __shfl_xor(tm, 8, 32));
      float mn = fmaxf(m[r], tm);
      alpha[r] = __expf(m[r] - mn);
      m[r] = mn;
      p0[r] = __expf(a0 - mn);
      p1[r] = __expf(a1 - mn);
      float rs = p0[r] + p1[r];
      rs += __shfl_xor(rs, 1, 32);
      rs += __shfl_xor(rs, 2, 32);
      rs += __shfl_xor(rs, 4, 32);
      rs += __shfl_xor(rs, 8, 32);
      lsum[r] = lsum[r] * alpha[r] + rs;
    }
#pragma unroll
    for (int r = 0; r < 8; ++r) { c0[r] *= alpha[r]; c1[r] *= alpha[r]; }

    // P: C layout -> LDS -> A layout
#pragma unroll
    for (int r = 0; r < 8; ++r) {
      pw[(r + 8 * hi) * 32 + n16]      = (_Float16)p0[r];
      pw[(r + 8 * hi) * 32 + 16 + n16] = (_Float16)p1[r];
    }
    asm volatile("s_wait_dscnt 0x0" ::: "memory");   // same-wave cross-lane LDS RAW
    const _Float16* prow = pw + n16 * 32;
    v16h Ap = ld_contig16(prow + selB);

    c0 = __builtin_amdgcn_wmma_f32_16x16x32_f16(false, Ap, false, Bs0, (short)0, c0, false, false);
    c1 = __builtin_amdgcn_wmma_f32_16x16x32_f16(false, Ap, false, Bs1, (short)0, c1, false, false);

    buf ^= 1;
  }

  // epilogue: out[h][q][d] = V[q][h*32+d] - O/l
#pragma unroll
  for (int r = 0; r < 8; ++r) {
    int q = q0 + r + 8 * hi;
    float inv = 1.0f / lsum[r];
    const float* vp = Vf + (size_t)q * EMBED + h * HEAD_DIM;
    float* op = out + (size_t)h * N_NODES * HEAD_DIM + (size_t)q * HEAD_DIM;
    op[n16]      = vp[n16]      - c0[r] * inv;
    op[16 + n16] = vp[16 + n16] - c1[r] * inv;
  }
}

// ---------- host launch --------------------------------------------------

extern "C" void kernel_launch(void* const* d_in, const int* in_sizes, int n_in,
                              void* d_out, int out_size, void* d_ws, size_t ws_size,
                              hipStream_t stream) {
  const float* x   = (const float*)d_in[0];
  const int*   adj = (const int*)  d_in[1];
  const float* Wq  = (const float*)d_in[2];
  const float* bq  = (const float*)d_in[3];
  const float* Wk  = (const float*)d_in[4];
  const float* bk  = (const float*)d_in[5];
  const float* Wv  = (const float*)d_in[6];
  const float* bv  = (const float*)d_in[7];
  float* out = (float*)d_out;

  // workspace carve-up (~14.4 MB total)
  char* ws = (char*)d_ws;
  size_t off = 0;
  auto carve = [&](size_t bytes) {
    size_t o = off;
    off = (off + bytes + 255) & ~(size_t)255;
    return o;
  };
  _Float16* xh   = (_Float16*)(ws + carve((size_t)N_NODES * IN_FEATS * 2));
  _Float16* Wqh  = (_Float16*)(ws + carve((size_t)EMBED * IN_FEATS * 2));
  _Float16* Wkh  = (_Float16*)(ws + carve((size_t)EMBED * IN_FEATS * 2));
  _Float16* Wvh  = (_Float16*)(ws + carve((size_t)EMBED * IN_FEATS * 2));
  _Float16* Qh   = (_Float16*)(ws + carve((size_t)N_NODES * EMBED * 2));
  _Float16* Kh   = (_Float16*)(ws + carve((size_t)N_NODES * EMBED * 2));
  float*    Vf   = (float*)   (ws + carve((size_t)N_NODES * EMBED * 4));
  _Float16* SVT  = (_Float16*)(ws + carve((size_t)HEADS * HEAD_DIM * N_NODES * 2));
  unsigned* bits = (unsigned*)(ws + carve((size_t)N_NODES * (N_NODES / 32) * 4));

  // 1) f32 -> f16 converts
  cvt_f16_kernel<<<(N_NODES * IN_FEATS + 255) / 256, 256, 0, stream>>>(x, xh, N_NODES * IN_FEATS);
  cvt_f16_kernel<<<(EMBED * IN_FEATS + 255) / 256, 256, 0, stream>>>(Wq, Wqh, EMBED * IN_FEATS);
  cvt_f16_kernel<<<(EMBED * IN_FEATS + 255) / 256, 256, 0, stream>>>(Wk, Wkh, EMBED * IN_FEATS);
  cvt_f16_kernel<<<(EMBED * IN_FEATS + 255) / 256, 256, 0, stream>>>(Wv, Wvh, EMBED * IN_FEATS);

  // 2) adj -> 2MB bitmask (L2-resident across all heads)
  pack_adj_kernel<<<(N_NODES * 128) / 8, 256, 0, stream>>>(adj, bits);

  // 3) Q/K/V projections via WMMA
  proj_kernel<<<dim3(N_NODES / 16, 3), 256, 0, stream>>>(xh, Wqh, Wkh, Wvh,
                                                         bq, bk, bv, Qh, Kh, Vf);

  // 4) softmax(V) over head_dim, transposed f16
  svt_kernel<<<(N_NODES * HEADS + 255) / 256, 256, 0, stream>>>(Vf, SVT);

  // 5) fused masked attention (async-LDS staged tiles)
  attn_kernel<<<dim3(N_NODES / 128, HEADS), 256, 0, stream>>>(Qh, Kh, SVT, Vf, bits, out);

  (void)in_sizes; (void)n_in; (void)out_size; (void)ws_size;
}
